// DenseSNN_76115410419938
// MI455X (gfx1250) — compile-verified, matrix operand order checked by reading, not run on hardware
//
#include <hip/hip_runtime.h>

// ---------------------------------------------------------------------------
// DenseSNN for MI455X (gfx1250): bf16 WMMA fused GEMM + LIF per layer/timestep
//   - double-buffered BK=64 LDS panels, async global->LDS staging (ASYNCcnt)
// ---------------------------------------------------------------------------

typedef __bf16 bf16_t;
typedef bf16_t v8bf  __attribute__((ext_vector_type(8)));
typedef bf16_t v16bf __attribute__((ext_vector_type(16)));
typedef float  v8f   __attribute__((ext_vector_type(8)));
typedef int    v4i   __attribute__((ext_vector_type(4)));

typedef __attribute__((address_space(1))) v4i* gv4i_p;   // global v4i*
typedef __attribute__((address_space(3))) v4i* lv4i_p;   // LDS v4i*

#define LDS_STRIDE 72   // 64 data halfs + 8 pad halfs (144B row) -> conflict-free b128 reads

#if __has_builtin(__builtin_amdgcn_s_wait_asynccnt)
#define WAIT_ASYNC(n) __builtin_amdgcn_s_wait_asynccnt(n)
#else
#define WAIT_ASYNC(n) asm volatile("s_wait_asynccnt %0" :: "i"(n) : "memory")
#endif

#if __has_builtin(__builtin_amdgcn_global_load_async_to_lds_b128)
__device__ __forceinline__ void async_cp16(const bf16_t* g, bf16_t* l) {
    __builtin_amdgcn_global_load_async_to_lds_b128(
        (gv4i_p)(g), (lv4i_p)(l), 0, 0);
}
#else
__device__ __forceinline__ void async_cp16(const bf16_t* g, bf16_t* l) {
    *(v8bf*)l = *(const v8bf*)g;    // sync fallback: load + ds_store
}
#endif

// -------- weight conversion: f32 [rows_src][K] -> bf16 [rows_dst][K], zero-pad
__global__ __launch_bounds__(256)
void k_convert_w(const float* __restrict__ src, bf16_t* __restrict__ dst,
                 int rows_src, int rows_dst, int K) {
    long i = (long)blockIdx.x * blockDim.x + threadIdx.x;
    long n = (long)rows_dst * K;
    if (i >= n) return;
    int r = (int)(i / K);
    dst[i] = (r < rows_src) ? (bf16_t)src[i] : (bf16_t)0.0f;
}

__global__ __launch_bounds__(256)
void k_zero_f32(float* __restrict__ p, long n) {
    long i = (long)blockIdx.x * blockDim.x + threadIdx.x;
    if (i < n) p[i] = 0.0f;
}

__global__ __launch_bounds__(256)
void k_pad_bias(const float* __restrict__ src, float* __restrict__ dst,
                int n_src, int n_dst) {
    int i = blockIdx.x * blockDim.x + threadIdx.x;
    if (i < n_dst) dst[i] = (i < n_src) ? src[i] : 0.0f;
}

__global__ __launch_bounds__(256)
void k_copy_out(const float* __restrict__ acc, float* __restrict__ out) {
    int i = blockIdx.x * blockDim.x + threadIdx.x;   // 128*1000
    if (i >= 128 * 1000) return;
    int b = i / 1000, o = i - b * 1000;
    out[i] = acc[b * 1024 + o];
}

// ---------------------------------------------------------------------------
// Fused layer kernel:  C[128,N] = A[128,K] @ W[N,K]^T + bias, then LIF update.
//   Block: 256 threads (8 waves). Block computes a 128 x 128 output tile.
//   Wave (wm in 0..3, wn in 0..1) computes 32 x 64 = 2x4 WMMA tiles.
//   Double-buffered BK=64 LDS K-panels, async global->LDS staging,
//   last pipeline stage peeled for a branch-free steady-state loop.
// ---------------------------------------------------------------------------
template<bool A_F32, bool OUT_ACC>
__global__ __launch_bounds__(256)
void k_lif_layer(const void* __restrict__ a_ptr,       // [128][K] f32 or bf16
                 const bf16_t* __restrict__ w,         // [N][K] bf16 (row-major)
                 const float* __restrict__ bias,       // [N] (pre-padded)
                 const float* __restrict__ beta_p,
                 const float* __restrict__ th_p,
                 float* __restrict__ memb,             // [128][N]
                 bf16_t* __restrict__ spk,             // [128][N]   (!OUT_ACC)
                 float* __restrict__ accp,             // [128][N]   (OUT_ACC)
                 int K, int N)
{
    __shared__ bf16_t sA[2][128 * LDS_STRIDE];
    __shared__ bf16_t sB[2][128 * LDS_STRIDE];

    const int tid  = threadIdx.x;
    const int lane = tid & 31;
    const int wave = tid >> 5;
    const int hl   = lane & 15;     // half-lane index
    const int hi   = lane >> 4;     // 0 / 1
    const int wm   = wave >> 1;     // 0..3 -> 32 output rows each
    const int wn   = wave & 1;      // 0..1 -> 64 output cols each
    const int nb   = blockIdx.x * 128;

    const float*  af = (const float*)a_ptr;
    const bf16_t* ab = (const bf16_t*)a_ptr;

    v8f acc[2][4];
    #pragma unroll
    for (int mt = 0; mt < 2; ++mt)
        #pragma unroll
        for (int nt = 0; nt < 4; ++nt)
            #pragma unroll
            for (int j = 0; j < 8; ++j)
                acc[mt][nt][j] = 0.0f;

    // ---- stage 64-wide K-panel kb into buffer buf (per thread: 4 A + 4 B chunks)
    auto stage = [&](int buf, int kb) {
        #pragma unroll
        for (int it = 0; it < 4; ++it) {
            int c   = tid + it * 256;        // 0..1023
            int row = c >> 3;
            int ck  = (c & 7) * 8;           // 0,8,...,56
            int gk  = kb + ck;
            if (A_F32) {                     // f32 -> bf16 convert while staging (sync)
                const float* s = af + row * K + gk;
                bf16_t* da = &sA[buf][row * LDS_STRIDE + ck];
                #pragma unroll
                for (int j = 0; j < 8; ++j) da[j] = (bf16_t)s[j];
            } else {
                async_cp16(ab + row * K + gk, &sA[buf][row * LDS_STRIDE + ck]);
            }
            async_cp16(w + (nb + row) * K + gk, &sB[buf][row * LDS_STRIDE + ck]);
        }
    };

    // ---- consume buffer cur: 2 k-chunks x (2x4 tiles) = 16 WMMAs
    auto compute = [&](int cur) {
        #pragma unroll
        for (int kc = 0; kc < 2; ++kc) {
            v16bf afr[2];
            #pragma unroll
            for (int mt = 0; mt < 2; ++mt) {
                const bf16_t* ap =
                    &sA[cur][(wm * 32 + mt * 16 + hl) * LDS_STRIDE + kc * 32];
                v8bf a0 = *(const v8bf*)(ap + hi * 8);        // K 0..7  / 8..15
                v8bf a1 = *(const v8bf*)(ap + 16 + hi * 8);   // K 16..23 / 24..31
                afr[mt] = __builtin_shufflevector(a0, a1,
                    0,1,2,3,4,5,6,7,8,9,10,11,12,13,14,15);
            }
            #pragma unroll
            for (int nt = 0; nt < 4; ++nt) {
                const bf16_t* bp =
                    &sB[cur][(wn * 64 + nt * 16 + hl) * LDS_STRIDE + kc * 32];
                v8bf b0 = *(const v8bf*)(bp + hi * 8);
                v8bf b1 = *(const v8bf*)(bp + 16 + hi * 8);
                v16bf bfr = __builtin_shufflevector(b0, b1,
                    0,1,2,3,4,5,6,7,8,9,10,11,12,13,14,15);
                #pragma unroll
                for (int mt = 0; mt < 2; ++mt) {
                    acc[mt][nt] = __builtin_amdgcn_wmma_f32_16x16x32_bf16(
                        false, afr[mt], false, bfr, (short)0, acc[mt][nt],
                        false, false);
                }
            }
        }
    };

    const int nst = K >> 6;                    // K/64 stages (>= 2 always)
    stage(0, 0);

    for (int s = 0; s < nst - 1; ++s) {        // branch-free steady state
        const int cur = s & 1;
        __syncthreads();                       // prev compute done -> safe to refill cur^1
        stage(cur ^ 1, (s + 1) << 6);          // overlap next panel fill with compute
        WAIT_ASYNC(A_F32 ? 4 : 8);             // drain down to next stage's in-flight loads
        __syncthreads();                       // all waves' panel-cur fills visible
        compute(cur);
    }
    __syncthreads();                           // peeled last stage
    WAIT_ASYNC(0);
    __syncthreads();
    compute((nst - 1) & 1);

    // ---- epilogue: bias + LIF (reset from previous mem, integrate, fire)
    float beta = fminf(fmaxf(beta_p[0], 0.0f), 1.0f);
    float th   = th_p[0];

    #pragma unroll
    for (int mt = 0; mt < 2; ++mt) {
        #pragma unroll
        for (int nt = 0; nt < 4; ++nt) {
            int gcol = nb + wn * 64 + nt * 16 + hl;
            float bi = bias[gcol];
            #pragma unroll
            for (int r = 0; r < 8; ++r) {
                int grow = wm * 32 + mt * 16 + hi * 8 + r;  // C/D layout: vgpr r, hi half
                int idx  = grow * N + gcol;
                float cur   = acc[mt][nt][r] + bi;
                float m     = memb[idx];
                float reset = (m - th > 0.0f) ? th : 0.0f;
                float mn    = beta * m + cur - reset;
                memb[idx]   = mn;
                float sv    = (mn - th > 0.0f) ? 1.0f : 0.0f;
                if (OUT_ACC) accp[idx] += sv;
                else         spk[idx]   = (bf16_t)sv;
            }
        }
    }
}

// ---------------------------------------------------------------------------
extern "C" void kernel_launch(void* const* d_in, const int* in_sizes, int n_in,
                              void* d_out, int out_size, void* d_ws, size_t ws_size,
                              hipStream_t stream) {
    const float* x   = (const float*)d_in[0];
    const float* w1  = (const float*)d_in[1];
    const float* b1  = (const float*)d_in[2];
    const float* w2  = (const float*)d_in[3];
    const float* b2  = (const float*)d_in[4];
    const float* w3  = (const float*)d_in[5];
    const float* b3  = (const float*)d_in[6];
    const float* wo  = (const float*)d_in[7];
    const float* bo  = (const float*)d_in[8];
    const float* beta1 = (const float*)d_in[9];
    const float* th1   = (const float*)d_in[10];
    const float* beta2 = (const float*)d_in[11];
    const float* th2   = (const float*)d_in[12];
    const float* beta3 = (const float*)d_in[13];
    const float* th3   = (const float*)d_in[14];
    const float* betao = (const float*)d_in[15];
    const float* tho   = (const float*)d_in[16];

    char* ws = (char*)d_ws;
    bf16_t* w1b  = (bf16_t*)(ws + (size_t)( 0u << 20));   // 2048x1024 bf16 (4MB)
    bf16_t* w2b  = (bf16_t*)(ws + (size_t)( 4u << 20));   // 2048x2048 bf16 (8MB)
    bf16_t* w3b  = (bf16_t*)(ws + (size_t)(12u << 20));   // 2048x2048 bf16 (8MB)
    bf16_t* wob  = (bf16_t*)(ws + (size_t)(20u << 20));   // 1024x2048 bf16 (4MB, pad)
    float*  m1   = (float*) (ws + (size_t)(24u << 20));   // 128x2048 f32
    float*  m2   = (float*) (ws + (size_t)(25u << 20));
    float*  m3   = (float*) (ws + (size_t)(26u << 20));
    float*  mo   = (float*) (ws + (size_t)(27u << 20));   // 128x1024 f32
    float*  accp = (float*) (ws + (size_t)(27u << 20) + (512u << 10));
    bf16_t* s1   = (bf16_t*)(ws + (size_t)(28u << 20));   // 128x2048 bf16
    bf16_t* s2   = (bf16_t*)(ws + (size_t)(28u << 20) + (512u << 10));
    bf16_t* s3   = (bf16_t*)(ws + (size_t)(29u << 20));
    float*  bop  = (float*) (ws + (size_t)(29u << 20) + (512u << 10)); // 1024 f32

    // one-time (per call) weight conversion + state init
    k_convert_w<<<(2048 * 1024 + 255) / 256, 256, 0, stream>>>(w1, w1b, 2048, 2048, 1024);
    k_convert_w<<<(2048 * 2048 + 255) / 256, 256, 0, stream>>>(w2, w2b, 2048, 2048, 2048);
    k_convert_w<<<(2048 * 2048 + 255) / 256, 256, 0, stream>>>(w3, w3b, 2048, 2048, 2048);
    k_convert_w<<<(1024 * 2048 + 255) / 256, 256, 0, stream>>>(wo, wob, 1000, 1024, 2048);
    k_pad_bias<<<4, 256, 0, stream>>>(bo, bop, 1000, 1024);
    // zero m1,m2,m3,mo,accp: contiguous 4MB region at ws+24MB
    k_zero_f32<<<((1u << 20) + 255) / 256, 256, 0, stream>>>(m1, (long)(1u << 20));

    // sequential scan over T=64 timesteps, 4 fused layer kernels per step
    for (int t = 0; t < 64; ++t) {
        const float* xt = x + (size_t)t * 128 * 1024;
        k_lif_layer<true,  false><<<16, 256, 0, stream>>>(
            xt, w1b, b1, beta1, th1, m1, s1, nullptr, 1024, 2048);
        k_lif_layer<false, false><<<16, 256, 0, stream>>>(
            s1, w2b, b2, beta2, th2, m2, s2, nullptr, 2048, 2048);
        k_lif_layer<false, false><<<16, 256, 0, stream>>>(
            s2, w3b, b3, beta3, th3, m3, s3, nullptr, 2048, 2048);
        k_lif_layer<false, true ><<< 8, 256, 0, stream>>>(
            s3, wob, bop, betao, tho, mo, nullptr, accp, 2048, 1024);
    }

    k_copy_out<<<(128 * 1000 + 255) / 256, 256, 0, stream>>>(accp, (float*)d_out);
}
